// MultiHeadAttention_29618094473500
// MI455X (gfx1250) — compile-verified, matrix operand order checked by reading, not run on hardware
//
#include <hip/hip_runtime.h>

// ---------------------------------------------------------------------------
// MI455X (gfx1250) multi-head attention pipeline, bf16 WMMA + f32 accumulate.
// ---------------------------------------------------------------------------

typedef __bf16 bf16_t;
typedef __bf16 v16bf __attribute__((ext_vector_type(16)));
typedef __bf16 v8bf  __attribute__((ext_vector_type(8)));
typedef float  v8f   __attribute__((ext_vector_type(8)));

#define DEV_INLINE __device__ __forceinline__

DEV_INLINE v8f vzero8() {
  v8f z;
#pragma unroll
  for (int i = 0; i < 8; ++i) z[i] = 0.0f;
  return z;
}

// Load a 16x32 bf16 tile (rows x K, row-major with leading dim `ld`) into the
// wave32 WMMA fragment layout for v_wmma_f32_16x16x32_bf16 (ISA 7.12.2):
//   lane L: row = L&15; elems[0..7] = K ((L>>4)*8 .. +7); elems[8..15] = +16.
// Works for A directly, and for B when the source is stored as B^T (row-major
// along the contraction dim) -> two contiguous 16-byte loads per lane.
DEV_INLINE v16bf load_tile_16x32(const bf16_t* base, int ld, int lane) {
  const int r  = lane & 15;
  const int k0 = (lane >> 4) << 3;
  const bf16_t* p = base + (long long)r * ld + k0;
  v8bf lo = *(const v8bf*)(p);
  v8bf hi = *(const v8bf*)(p + 16);
  v16bf out;
#pragma unroll
  for (int i = 0; i < 8; ++i) { out[i] = lo[i]; out[i + 8] = hi[i]; }
  return out;
}

DEV_INLINE v8f wmma_bf16f32(v16bf a, v16bf b, v8f c) {
  return __builtin_amdgcn_wmma_f32_16x16x32_bf16(false, a, false, b, (short)0, c,
                                                 false, false);
}

// ---------------------------------------------------------------------------
// Packing / conversion kernels
// ---------------------------------------------------------------------------

__global__ void f32_to_bf16_kernel(const float* __restrict__ in,
                                   bf16_t* __restrict__ out, int n) {
  int i = blockIdx.x * blockDim.x + threadIdx.x;
  if (i < n) out[i] = (bf16_t)in[i];
}

// x[b][c][hw] (2,256,4096) -> xT[b][hw][c]  (contiguous along c for B-frags)
__global__ void pack_xT_kernel(const float* __restrict__ x,
                               bf16_t* __restrict__ xT) {
  int i   = blockIdx.x * blockDim.x + threadIdx.x;  // 2*256*4096 threads
  int b   = i >> 20;
  int rem = i & 1048575;
  int c   = rem >> 12;
  int hw  = rem & 4095;
  xT[(((long long)b << 12) | hw) * 256 + c] = (bf16_t)x[i];
}

// im2col for the stride-2 2x2 conv: pat[b][hw2][c*4 + i*2 + j]
// (k-order matches Wkv's natural [o][c][i][j] flattening)
__global__ void pack_patches_kernel(const float* __restrict__ x,
                                    bf16_t* __restrict__ pat) {
  int i   = blockIdx.x * blockDim.x + threadIdx.x;  // 2*1024*1024 threads
  int b   = i >> 20;
  int rem = i & 1048575;
  int hw2 = rem >> 10;
  int k   = rem & 1023;
  int c   = k >> 2;
  int ij  = k & 3;
  int h2  = hw2 >> 5, w2 = hw2 & 31;
  int hs  = 2 * h2 + (ij >> 1), wsp = 2 * w2 + (ij & 1);
  pat[i] = (bf16_t)x[(((long long)(b * 256 + c)) * 64 + hs) * 64 + wsp];
}

// ---------------------------------------------------------------------------
// Generic bf16 WMMA GEMM:  C[M x N] = A[M x K] * B[K x N]   (K % 64 == 0)
//   A  : row-major [M][K]  (shared across batch — weights)
//   Bt : row-major [N][K]  (= B^T), per-batch stride btStride
// 128 threads = 4 waves arranged 2x2; block tile 64x64; wave tile 32(M)x32(N):
// per K-chunk each wave loads 2 A-frags + 2 B-frags and issues 4 WMMAs. The
// K-loop is unrolled x2 with ping-pong fragment buffers (no rotation copies):
// buffer Y's loads issue before buffer X's WMMAs and vice versa, so
// s_wait_loadcnt always retires behind a full WMMA group. Tail is peeled.
// mode 0: scatter to q_seq[b][h][n][d]
// mode 1: rows <512 -> k_seq[b][h][m][d]; rows >=512 -> vT[b][h][d][m]
// mode 2: f32 output out[b][o][hw]
// ---------------------------------------------------------------------------
__global__ void __launch_bounds__(128)
gemm_bf16_wmma(const bf16_t* __restrict__ A, const bf16_t* __restrict__ Bt,
               int M, int N, int K, long long btStride, int mode,
               bf16_t* __restrict__ outb, float* __restrict__ outf) {
  const int lane = threadIdx.x & 31;
  const int wid  = threadIdx.x >> 5;
  const int b    = blockIdx.z;
  const int m0   = blockIdx.y * 64 + (wid & 1) * 32;
  const int n0   = blockIdx.x * 64 + (wid >> 1) * 32;

  const bf16_t* Ap = A + (long long)m0 * K;
  const bf16_t* Bp = Bt + (long long)b * btStride + (long long)n0 * K;

  v8f acc[4];
#pragma unroll
  for (int t = 0; t < 4; ++t) acc[t] = vzero8();

  v16bf xa0, xa1, xb0, xb1;  // ping buffer
  v16bf ya0, ya1, yb0, yb1;  // pong buffer

  auto loadX = [&](int kk) {
    xa0 = load_tile_16x32(Ap + kk, K, lane);
    xa1 = load_tile_16x32(Ap + (long long)16 * K + kk, K, lane);
    xb0 = load_tile_16x32(Bp + kk, K, lane);
    xb1 = load_tile_16x32(Bp + (long long)16 * K + kk, K, lane);
  };
  auto loadY = [&](int kk) {
    ya0 = load_tile_16x32(Ap + kk, K, lane);
    ya1 = load_tile_16x32(Ap + (long long)16 * K + kk, K, lane);
    yb0 = load_tile_16x32(Bp + kk, K, lane);
    yb1 = load_tile_16x32(Bp + (long long)16 * K + kk, K, lane);
  };
  auto mmaX = [&]() {
    acc[0] = wmma_bf16f32(xa0, xb0, acc[0]);
    acc[1] = wmma_bf16f32(xa0, xb1, acc[1]);
    acc[2] = wmma_bf16f32(xa1, xb0, acc[2]);
    acc[3] = wmma_bf16f32(xa1, xb1, acc[3]);
  };
  auto mmaY = [&]() {
    acc[0] = wmma_bf16f32(ya0, yb0, acc[0]);
    acc[1] = wmma_bf16f32(ya0, yb1, acc[1]);
    acc[2] = wmma_bf16f32(ya1, yb0, acc[2]);
    acc[3] = wmma_bf16f32(ya1, yb1, acc[3]);
  };

  loadX(0);
  int kc = 0;
  for (; kc + 64 < K; kc += 64) {
    __builtin_prefetch(Ap + kc + 64, 0, 1);  // global_prefetch_b8
    loadY(kc + 32);
    mmaX();
    loadX(kc + 64);
    mmaY();
  }
  // Tail: kc == K - 64 (K is a multiple of 64).
  loadY(kc + 32);
  mmaX();
  mmaY();

  // D layout: lane holds column n_local = lane&15; VGPR v holds row v (+8 for
  // upper lane half). Tile t: rows m0 + (t>>1)*16, cols n0 + (t&1)*16.
  const int col = lane & 15;
  const int rof = (lane >> 4) << 3;
#pragma unroll
  for (int t = 0; t < 4; ++t) {
#pragma unroll
    for (int v = 0; v < 8; ++v) {
      float val = acc[t][v];
      int o = m0 + (t >> 1) * 16 + rof + v;
      int n = n0 + (t & 1) * 16 + col;
      if (mode == 0) {
        outb[(((long long)(b * 8 + (o >> 6))) * 4096 + n) * 64 + (o & 63)] =
            (bf16_t)val;
      } else if (mode == 1) {
        if (o < 512) {
          outb[(((long long)(b * 8 + (o >> 6))) * 1024 + n) * 64 + (o & 63)] =
              (bf16_t)val;
        } else {
          int o2 = o - 512;
          outb[1048576ll +
               (((long long)(b * 8 + (o2 >> 6))) * 64 + (o2 & 63)) * 1024 + n] =
              (bf16_t)val;
        }
      } else {
        outf[((long long)(b * 256 + o)) * 4096 + n] = val;
      }
    }
  }
}

// ---------------------------------------------------------------------------
// Flash-attention: each wave owns one 16-row query tile (N=4096, M=1024, d=64).
// scores = (Q K^T) * 8   (reference multiplies by sqrt(DK))
// Online softmax over 32-key chunks, two chunks per iteration with ping-pong
// K-fragment buffers; P transposed through LDS for the PV GEMM.
// ---------------------------------------------------------------------------
__global__ void __launch_bounds__(128)
attn_flash_wmma(const bf16_t* __restrict__ q_seq, const bf16_t* __restrict__ k_seq,
                const bf16_t* __restrict__ vT, bf16_t* __restrict__ attn_seq) {
  __shared__ __align__(16) bf16_t Plds[4][16][32];

  const int lane = threadIdx.x & 31;
  const int wid  = threadIdx.x >> 5;
  const int g    = blockIdx.x * 4 + wid;   // 4096 waves total
  const int b    = g >> 11;
  const int rem  = g & 2047;
  const int h    = rem >> 8;
  const int n0   = (rem & 255) * 16;

  const bf16_t* qb = q_seq + (((long long)(b * 8 + h)) * 4096 + n0) * 64;
  const v16bf qa0 = load_tile_16x32(qb, 64, lane);       // d = 0..31
  const v16bf qa1 = load_tile_16x32(qb + 32, 64, lane);  // d = 32..63

  const bf16_t* kb = k_seq + ((long long)(b * 8 + h)) * 1024 * 64;
  const bf16_t* vb = vT + ((long long)(b * 8 + h)) * 64 * 1024;

  float mrow[8], lrow[8];
#pragma unroll
  for (int v = 0; v < 8; ++v) { mrow[v] = -__builtin_inff(); lrow[v] = 0.0f; }
  v8f acc[4];
#pragma unroll
  for (int t = 0; t < 4; ++t) acc[t] = vzero8();

  const int col = lane & 15;
  const int rof = (lane >> 4) << 3;

  v16bf kA0, kA1, kA2, kA3;  // ping K-fragment buffer (32 keys)
  v16bf kB0, kB1, kB2, kB3;  // pong K-fragment buffer

  auto loadKA = [&](int m) {
    kA0 = load_tile_16x32(kb + (long long)m * 64, 64, lane);
    kA1 = load_tile_16x32(kb + (long long)m * 64 + 32, 64, lane);
    kA2 = load_tile_16x32(kb + (long long)(m + 16) * 64, 64, lane);
    kA3 = load_tile_16x32(kb + (long long)(m + 16) * 64 + 32, 64, lane);
  };
  auto loadKB = [&](int m) {
    kB0 = load_tile_16x32(kb + (long long)m * 64, 64, lane);
    kB1 = load_tile_16x32(kb + (long long)m * 64 + 32, 64, lane);
    kB2 = load_tile_16x32(kb + (long long)(m + 16) * 64, 64, lane);
    kB3 = load_tile_16x32(kb + (long long)(m + 16) * 64 + 32, 64, lane);
  };

  // Process one 32-key chunk whose K-fragments are already resident.
  auto step = [&](const v16bf& f0, const v16bf& f1, const v16bf& f2,
                  const v16bf& f3, int mc) {
    v8f s0 = wmma_bf16f32(qa0, f0, vzero8());
    s0     = wmma_bf16f32(qa1, f1, s0);
    v8f s1 = wmma_bf16f32(qa0, f2, vzero8());
    s1     = wmma_bf16f32(qa1, f3, s1);

    float mnew[8], alpha[8];
#pragma unroll
    for (int v = 0; v < 8; ++v) {
      s0[v] *= 8.0f;  // * sqrt(DK), per reference
      s1[v] *= 8.0f;
      float c = fmaxf(s0[v], s1[v]);
#pragma unroll
      for (int off = 1; off < 16; off <<= 1)   // stays within 16-lane half
        c = fmaxf(c, __shfl_xor(c, off, 32));
      mnew[v]  = fmaxf(mrow[v], c);
      alpha[v] = __expf(mrow[v] - mnew[v]);
      mrow[v]  = mnew[v];
    }
#pragma unroll
    for (int v = 0; v < 8; ++v) {
      s0[v] = __expf(s0[v] - mnew[v]);
      s1[v] = __expf(s1[v] - mnew[v]);
      float r = s0[v] + s1[v];
#pragma unroll
      for (int off = 1; off < 16; off <<= 1)
        r += __shfl_xor(r, off, 32);
      lrow[v] = lrow[v] * alpha[v] + r;
    }

    // Transpose P (D layout: lane=column) -> A layout via LDS. Per-wave DS
    // ops are in-order on CDNA5 (no barrier needed within a wave); stores
    // issue before the V-fragment global loads so DS latency overlaps VMEM.
#pragma unroll
    for (int v = 0; v < 8; ++v) {
      Plds[wid][rof + v][col]      = (bf16_t)s0[v];
      Plds[wid][rof + v][col + 16] = (bf16_t)s1[v];
    }

    // PV: A = P (16 x 32 over m), B = V chunk (32 x 16 per d-tile), from
    // vT[d][m] so B-frag loads are contiguous along m.
    v16bf vf0 = load_tile_16x32(vb + (long long)0  * 1024 + mc, 1024, lane);
    v16bf vf1 = load_tile_16x32(vb + (long long)16 * 1024 + mc, 1024, lane);
    v16bf vf2 = load_tile_16x32(vb + (long long)32 * 1024 + mc, 1024, lane);
    v16bf vf3 = load_tile_16x32(vb + (long long)48 * 1024 + mc, 1024, lane);

#pragma unroll
    for (int t = 0; t < 4; ++t)
#pragma unroll
      for (int v = 0; v < 8; ++v) acc[t][v] *= alpha[v];

    v16bf pa = load_tile_16x32(&Plds[wid][0][0], 32, lane);
    acc[0] = wmma_bf16f32(pa, vf0, acc[0]);
    acc[1] = wmma_bf16f32(pa, vf1, acc[1]);
    acc[2] = wmma_bf16f32(pa, vf2, acc[2]);
    acc[3] = wmma_bf16f32(pa, vf3, acc[3]);
  };

  loadKA(0);
  int mc = 0;
  for (; mc + 64 < 1024; mc += 64) {
    __builtin_prefetch(kb + (long long)(mc + 64) * 64, 0, 1);
    loadKB(mc + 32);
    step(kA0, kA1, kA2, kA3, mc);
    loadKA(mc + 64);
    step(kB0, kB1, kB2, kB3, mc + 32);
  }
  // Tail: mc == 960.
  loadKB(mc + 32);
  step(kA0, kA1, kA2, kA3, mc);
  step(kB0, kB1, kB2, kB3, mc + 32);

  // Epilogue: normalize and scatter to attn_seq[b][n][c] (c = h*64 + d,
  // contiguous along c for the projection GEMM's B-frag loads).
#pragma unroll
  for (int t = 0; t < 4; ++t) {
#pragma unroll
    for (int v = 0; v < 8; ++v) {
      float val = acc[t][v] / lrow[v];
      int n = n0 + rof + v;
      int c = h * 64 + t * 16 + col;
      attn_seq[((long long)b * 4096 + n) * 512 + c] = (bf16_t)val;
    }
  }
}

// ---------------------------------------------------------------------------
// Host-side orchestration
// ---------------------------------------------------------------------------
extern "C" void kernel_launch(void* const* d_in, const int* in_sizes, int n_in,
                              void* d_out, int out_size, void* d_ws, size_t ws_size,
                              hipStream_t stream) {
  (void)in_sizes; (void)n_in; (void)out_size; (void)ws_size;
  const float* x     = (const float*)d_in[0];  // (2,256,64,64)
  const float* Wq    = (const float*)d_in[1];  // (512,256)
  const float* Wkv   = (const float*)d_in[2];  // (1024,256,2,2)
  const float* Wproj = (const float*)d_in[3];  // (256,512)
  float* out = (float*)d_out;                  // (2,256,64,64)

  char* ws = (char*)d_ws;
  size_t off = 0;
  auto alloc_bf = [&](size_t elems) {
    bf16_t* p = (bf16_t*)(ws + off);
    off += ((elems * 2 + 255) / 256) * 256;
    return p;
  };
  bf16_t* wq_b     = alloc_bf(512 * 256);            // 0.25 MB
  bf16_t* wkv_b    = alloc_bf(1024 * 1024);          // 2 MB
  bf16_t* wproj_b  = alloc_bf(256 * 512);            // 0.25 MB
  bf16_t* xT       = alloc_bf(2ull * 4096 * 256);    // 4 MB
  bf16_t* pat      = alloc_bf(2ull * 1024 * 1024);   // 4 MB
  bf16_t* q_seq    = alloc_bf(2ull * 8 * 4096 * 64); // 8 MB
  bf16_t* kv_seq   = alloc_bf(2ull * 1048576);       // 4 MB: [k | vT]
  bf16_t* attn_seq = alloc_bf(2ull * 4096 * 512);    // 8 MB

  // 1) precision conversion + data layout packing
  f32_to_bf16_kernel<<<(512 * 256 + 255) / 256, 256, 0, stream>>>(Wq, wq_b, 512 * 256);
  f32_to_bf16_kernel<<<(1024 * 1024 + 255) / 256, 256, 0, stream>>>(Wkv, wkv_b, 1024 * 1024);
  f32_to_bf16_kernel<<<(256 * 512 + 255) / 256, 256, 0, stream>>>(Wproj, wproj_b, 256 * 512);
  pack_xT_kernel<<<2097152 / 256, 256, 0, stream>>>(x, xT);
  pack_patches_kernel<<<2097152 / 256, 256, 0, stream>>>(x, pat);

  // 2) q = Wq * x        : M=512, N=4096, K=256
  gemm_bf16_wmma<<<dim3(64, 8, 2), 128, 0, stream>>>(
      wq_b, xT, 512, 4096, 256, 4096ll * 256, 0, q_seq, nullptr);

  // 3) kv = Wkv (*) x    : M=1024, N=1024, K=1024 (im2col GEMM)
  gemm_bf16_wmma<<<dim3(16, 16, 2), 128, 0, stream>>>(
      wkv_b, pat, 1024, 1024, 1024, 1024ll * 1024, 1, kv_seq, nullptr);

  // 4) flash attention   : 4096 waves, one 16-row query tile each
  attn_flash_wmma<<<1024, 128, 0, stream>>>(q_seq, kv_seq, kv_seq + 1048576,
                                            attn_seq);

  // 5) out = Wproj * attn : M=256, N=4096, K=512 (fp32 output)
  gemm_bf16_wmma<<<dim3(64, 4, 2), 128, 0, stream>>>(
      wproj_b, attn_seq, 256, 4096, 512, 4096ll * 512, 2, nullptr, out);
}
